// QuantumLSTM_65481071400315
// MI455X (gfx1250) — matrix operand order repaired and checked by reading, not compiled, not run on hardware
//
#include <hip/hip_runtime.h>
#include <hip/hip_bf16.h>

#define T_STEPS 512
#define B_SZ    256
#define D_SZ    128
#define H_SZ    256
#define DH      384
#define NCOL    32          // 4 gates * Q
#define ROWS    16          // batch rows per block (WMMA M)
#define LDA     392         // padded LDS row stride (bf16 elems): 784B == 4 dwords mod 64 banks
#define NTHREADS 256

typedef __attribute__((ext_vector_type(16))) __bf16 v16bf;
typedef __attribute__((ext_vector_type(8)))  float  v8f;

__device__ __forceinline__ unsigned short f2bf(float f) {
    unsigned int u = __float_as_uint(f);
    u += 0x7FFFu + ((u >> 16) & 1u);           // round-to-nearest-even
    return (unsigned short)(u >> 16);
}
__device__ __forceinline__ float fast_tanh(float x) {
    float e = __expf(-2.0f * x);
    return (1.0f - e) / (1.0f + e);
}
__device__ __forceinline__ float fast_sig(float x) {
    return 1.0f / (1.0f + __expf(-x));
}

// Async 16B global->LDS copy, tracked by ASYNCcnt (CDNA5).
// LDS address = low 32 bits of the generic pointer (flat LDS aperture truncation).
__device__ __forceinline__ void async_ld16(void* lds, const void* gaddr) {
    unsigned lo = (unsigned)(uintptr_t)lds;
    asm volatile("global_load_async_to_lds_b128 %0, %1, off"
                 :: "v"(lo), "v"(gaddr) : "memory");
}
__device__ __forceinline__ void wait_async0() {
    asm volatile("s_wait_asynccnt 0x0" ::: "memory");
}

__global__ __launch_bounds__(NTHREADS)
void qlstm_scan(const float* __restrict__ x,
                const float* __restrict__ hx0, const float* __restrict__ cx0,
                const float* __restrict__ Wf, const float* __restrict__ bfv, const float* __restrict__ pf,
                const float* __restrict__ Wi, const float* __restrict__ biv, const float* __restrict__ pi,
                const float* __restrict__ Wu, const float* __restrict__ buv, const float* __restrict__ pu,
                const float* __restrict__ Wo, const float* __restrict__ bov, const float* __restrict__ po,
                float* __restrict__ out)
{
    __shared__ __align__(16) unsigned short sW[NCOL][LDA]; // [Wx | Wh] bf16 (32 x 384 used)
    __shared__ __align__(16) unsigned short sA[ROWS][LDA]; // [x_t | hx] bf16 (16 x 384 used)
    __shared__ __align__(16) float sXs[2][ROWS][D_SZ];     // async-staged x double buffer (f32)
    __shared__ float zbuf[ROWS][NCOL + 4];                 // z = comb @ W^T + b
    __shared__ float sGate[4][ROWS];                       // f, i, g, o per batch row
    __shared__ float sBias[NCOL];
    __shared__ float sP[NCOL];

    const int tid  = threadIdx.x;
    const int lane = tid & 31;             // wave32
    const int wv   = tid >> 5;             // 8 waves
    const int rowbase = blockIdx.x * ROWS;

    const float* Wl[4] = {Wf, Wi, Wu, Wo};
    const float* Bl[4] = {bfv, biv, buv, bov};
    const float* Pl[4] = {pf, pi, pu, po};

    // staging role: each thread owns 8 consecutive floats of one x row,
    // and reads back exactly the same 8 floats -> per-wave ASYNCcnt wait suffices
    const int xr  = tid >> 4;
    const int xkb = (tid & 15) * 8;

    // ---- one-time: weights/bias/params -> LDS (bf16 weights) ----
    for (int idx = tid; idx < NCOL * DH; idx += NTHREADS) {
        int n = idx / DH, k = idx - n * DH;
        sW[n][k] = f2bf(Wl[n >> 3][(n & 7) * DH + k]);
    }
    if (tid < NCOL) {
        sBias[tid] = Bl[tid >> 3][tid & 7];
        sP[tid]    = Pl[tid >> 3][tid & 7];
    }
    // hx0 -> sA columns [128, 384)
    for (int idx = tid; idx < ROWS * H_SZ; idx += NTHREADS) {
        int r = idx >> 8, k = idx & 255;
        sA[r][D_SZ + k] = f2bf(hx0[(size_t)(rowbase + r) * H_SZ + k]);
    }
    // cx -> registers, 16 cells per thread
    const int blr = tid >> 4;              // local batch row 0..15
    const int hb  = (tid & 15) * 16;       // hidden base
    float cx[16];
    #pragma unroll
    for (int j = 0; j < 16; ++j)
        cx[j] = cx0[(size_t)(rowbase + blr) * H_SZ + hb + j];

    // kick off async staging of x_0
    {
        const float* g0 = x + ((size_t)(rowbase + xr)) * D_SZ + xkb;
        async_ld16(&sXs[0][xr][xkb],     g0);
        async_ld16(&sXs[0][xr][xkb + 4], g0 + 4);
    }

    __syncthreads();

    // GEMM role per wave: 2 N-tiles x 4 K-slices (3 K-blocks each)
    const int ntile  = wv & 1;
    const int kslice = wv >> 1;
    const int arow   = lane & 15;
    const int sel    = lane >> 4;

    for (int t = 0; t < T_STEPS; ++t) {
        // ---- phase A: consume async-staged x_t (f32 -> bf16), restage x_{t+1},
        //               init z with bias ----
        {
            wait_async0();   // staged data for step t is in LDS
            const float* sp = &sXs[t & 1][xr][xkb];
            float4 v0 = ((const float4*)sp)[0];
            float4 v1 = ((const float4*)sp)[1];
            unsigned short tmp[8] = {f2bf(v0.x), f2bf(v0.y), f2bf(v0.z), f2bf(v0.w),
                                     f2bf(v1.x), f2bf(v1.y), f2bf(v1.z), f2bf(v1.w)};
            *(uint4*)&sA[xr][xkb] = *(const uint4*)tmp;
            if (t + 1 < T_STEPS) {
                const float* gn = x + ((size_t)(t + 1) * B_SZ + rowbase + xr) * D_SZ + xkb;
                async_ld16(&sXs[(t + 1) & 1][xr][xkb],     gn);
                async_ld16(&sXs[(t + 1) & 1][xr][xkb + 4], gn + 4);
            }
        }
        for (int idx = tid; idx < ROWS * NCOL; idx += NTHREADS)
            zbuf[idx >> 5][idx & 31] = sBias[idx & 31];
        __syncthreads();

        // ---- phase B: z += [x_t | hx] @ W^T via WMMA bf16, K-split + LDS add ----
        {
            v8f acc = {};
            #pragma unroll
            for (int kk = 0; kk < 3; ++kk) {
                const int kb0 = (kslice * 3 + kk) * 32;
                union { uint4 q[2]; v16bf v; } ua, ub;
                // A fragment (16x32 bf16): lanes 0-15 rows, K pairs per ISA layout
                ua.q[0] = *(const uint4*)&sA[arow][kb0 + sel * 8];
                ua.q[1] = *(const uint4*)&sA[arow][kb0 + 16 + sel * 8];
                // B fragment (32x16 bf16): lane = N col, K pairs per ISA layout (same as A)
                ub.q[0] = *(const uint4*)&sW[ntile * 16 + arow][kb0 + sel * 8];
                ub.q[1] = *(const uint4*)&sW[ntile * 16 + arow][kb0 + 16 + sel * 8];
                acc = __builtin_amdgcn_wmma_f32_16x16x32_bf16(
                        false, ua.v, false, ub.v, (short)0, acc, false, false);
            }
            #pragma unroll
            for (int j = 0; j < 8; ++j)
                atomicAdd(&zbuf[j + sel * 8][ntile * 16 + arow], acc[j]);
        }
        __syncthreads();

        // ---- phase C: quantum readout  prod_{q=1..7} cos(z_q + p_q) -> gate ----
        if (tid < 64) {
            int b = tid >> 2, g = tid & 3;
            float prod = 1.0f;
            #pragma unroll
            for (int q = 1; q < 8; ++q)
                prod *= __cosf(zbuf[b][g * 8 + q] + sP[g * 8 + q]);
            sGate[g][b] = (g == 2) ? fast_tanh(prod) : fast_sig(prod);
        }
        __syncthreads();

        // ---- phase D: cell update, hx -> LDS(bf16) + out(f32) ----
        {
            float fg = sGate[0][blr], ig = sGate[1][blr];
            float gg = sGate[2][blr], og = sGate[3][blr];
            float igg = ig * gg;
            float hv[16];
            unsigned short hb16[16];
            #pragma unroll
            for (int j = 0; j < 16; ++j) {
                float c = fmaf(fg, cx[j], igg);
                cx[j] = c;
                float h = og * fast_tanh(c);
                hv[j] = h;
                hb16[j] = f2bf(h);
            }
            *(uint4*)&sA[blr][D_SZ + hb]     = *(const uint4*)&hb16[0];
            *(uint4*)&sA[blr][D_SZ + hb + 8] = *(const uint4*)&hb16[8];
            float* op = out + ((size_t)t * B_SZ + rowbase + blr) * H_SZ + hb;
            ((float4*)op)[0] = *(float4*)&hv[0];
            ((float4*)op)[1] = *(float4*)&hv[4];
            ((float4*)op)[2] = *(float4*)&hv[8];
            ((float4*)op)[3] = *(float4*)&hv[12];
        }
        __syncthreads();
    }

    // ---- final (hx, cx) appended after outputs ----
    {
        size_t base = (size_t)T_STEPS * B_SZ * H_SZ;
        float* hxo = out + base + (size_t)(rowbase + blr) * H_SZ + hb;
        float* cxo = hxo + (size_t)B_SZ * H_SZ;
        #pragma unroll
        for (int j = 0; j < 16; ++j) {
            unsigned short u = sA[blr][D_SZ + hb + j];
            hxo[j] = __uint_as_float((unsigned int)u << 16);
            cxo[j] = cx[j];
        }
    }
}

extern "C" void kernel_launch(void* const* d_in, const int* in_sizes, int n_in,
                              void* d_out, int out_size, void* d_ws, size_t ws_size,
                              hipStream_t stream) {
    const float* X   = (const float*)d_in[0];
    const float* hx0 = (const float*)d_in[1];
    const float* cx0 = (const float*)d_in[2];
    const float* Wf  = (const float*)d_in[3];
    const float* bfv = (const float*)d_in[4];
    const float* pf  = (const float*)d_in[5];
    const float* Wi  = (const float*)d_in[6];
    const float* biv = (const float*)d_in[7];
    const float* pi  = (const float*)d_in[8];
    const float* Wu  = (const float*)d_in[9];
    const float* buv = (const float*)d_in[10];
    const float* pu  = (const float*)d_in[11];
    const float* Wo  = (const float*)d_in[12];
    const float* bov = (const float*)d_in[13];
    const float* po  = (const float*)d_in[14];
    float* out = (float*)d_out;

    qlstm_scan<<<B_SZ / ROWS, NTHREADS, 0, stream>>>(
        X, hx0, cx0, Wf, bfv, pf, Wi, biv, pi, Wu, buv, pu, Wo, bov, po, out);
}
